// EMO_13907104105094
// MI455X (gfx1250) — compile-verified, matrix operand order checked by reading, not run on hardware
//
#include <hip/hip_runtime.h>
#include <hip/hip_bf16.h>

typedef __attribute__((ext_vector_type(16))) __bf16 v16bf;
typedef __attribute__((ext_vector_type(8)))  float  v8f;
typedef unsigned short ushort_t;

#define NPOS 49
#define MPAD 64
#define NWIN 1024
#define MTOT (NWIN * NPOS)   // 50176

__device__ inline ushort_t bfbits(float f) {
    unsigned u = __builtin_bit_cast(unsigned, f);
    return (ushort_t)((u + 0x7FFFu + ((u >> 16) & 1u)) >> 16);
}
__device__ inline float bf2f(ushort_t s) {
    unsigned u = ((unsigned)s) << 16;
    return __builtin_bit_cast(float, u);
}

// ---- gfx1250 async global->LDS copy (ASYNCcnt-tracked), 16B per lane ----
__device__ inline void async_g2l_b128(unsigned lds_off, const void* gaddr) {
    asm volatile("global_load_async_to_lds_b128 %0, %1, off"
                 :: "v"(lds_off), "v"(gaddr) : "memory");
}
__device__ inline void async_wait0() {
    asm volatile("s_wait_asynccnt 0x0" ::: "memory");
}

// ---- WMMA fragment helpers (gfx1250 wave32 layouts, ISA 7.12.2) ----
__device__ inline v16bf load_a_frag(const ushort_t* lds, int mbase, int kbase, int ldk) {
    int lane = threadIdx.x & 31;
    int m = mbase + (lane & 15);
    int grp = lane >> 4;
    v16bf a;
#pragma unroll
    for (int e = 0; e < 16; ++e) {
        int v = e >> 1, p = e & 1;
        int k = kbase + ((v >= 4) ? 16 : 0) + grp * 8 + (v & 3) * 2 + p;
        a[e] = __builtin_bit_cast(__bf16, lds[m * ldk + k]);
    }
    return a;
}
// B where LDS holds the operand as [n][k] (weight-transposed layout)
__device__ inline v16bf load_b_frag_t(const ushort_t* lds, int nbase, int kbase, int ldk) {
    int lane = threadIdx.x & 31;
    int n = nbase + (lane & 15);
    int grp = lane >> 4;
    v16bf b;
#pragma unroll
    for (int e = 0; e < 16; ++e) {
        int v = e >> 1, p = e & 1;
        int k = kbase + ((v >= 4) ? 16 : 0) + grp * 8 + (v & 3) * 2 + p;
        b[e] = __builtin_bit_cast(__bf16, lds[n * ldk + k]);
    }
    return b;
}
// B where LDS holds the operand as [k][n] (natural row-major)
__device__ inline v16bf load_b_frag_n(const ushort_t* lds, int nbase, int kbase, int ldn) {
    int lane = threadIdx.x & 31;
    int n = nbase + (lane & 15);
    int grp = lane >> 4;
    v16bf b;
#pragma unroll
    for (int e = 0; e < 16; ++e) {
        int v = e >> 1, p = e & 1;
        int k = kbase + ((v >= 4) ? 16 : 0) + grp * 8 + (v & 3) * 2 + p;
        b[e] = __builtin_bit_cast(__bf16, lds[k * ldn + n]);
    }
    return b;
}
__device__ inline v8f wmma_bf16(v16bf a, v16bf b, v8f c) {
    return __builtin_amdgcn_wmma_f32_16x16x32_bf16(false, a, false, b, (short)0, c, false, false);
}
__device__ inline float gelu_exact(float v) {
    return 0.5f * v * (1.0f + erff(v * 0.70710678118654752f));
}

// ---- Kernel 0a: f32 -> bf16 weight conversion (run once per launch) ----
__global__ __launch_bounds__(256) void k_cvt(const float* __restrict__ src,
                                             ushort_t* __restrict__ dst, int n) {
    int i = blockIdx.x * 256 + threadIdx.x;
    if (i < n) dst[i] = bfbits(src[i]);
}

// ---- Kernel 0b: zero the pad rows (49..63) of the padded proj input ----
__global__ __launch_bounds__(256) void k_zero_pad(ushort_t* __restrict__ x2) {
    const int CHUNKS = NWIN * 15 * 128;         // 16B chunks
    int idx = blockIdx.x * 256 + threadIdx.x;
    if (idx >= CHUNKS) return;
    int win = idx / (15 * 128);
    int rem = idx - win * (15 * 128);
    int r = NPOS + rem / 128;
    int c = (rem & 127) * 8;
    uint4* p = (uint4*)(x2 + ((size_t)win * MPAD + r) * 1024 + c);
    *p = make_uint4(0u, 0u, 0u, 0u);
}

// ---- Kernel 1: dilated window partition + channel-LN -> bf16 [win][64][256]
__global__ __launch_bounds__(256) void k_ln(const float* __restrict__ x,
                                            const float* __restrict__ pre_w,
                                            const float* __restrict__ pre_b,
                                            ushort_t* __restrict__ xln) {
    int win = blockIdx.x;
    int imgB = win >> 4, wIdx = win & 15;
    int wy = wIdx >> 2, wx = wIdx & 3;
    int wave = threadIdx.x >> 5, lane = threadIdx.x & 31;
    for (int p = wave; p < NPOS; p += 8) {
        int py = p / 7, px = p % 7;
        int h = py * 4 + wy, w = px * 4 + wx;
        const float* xb = x + (size_t)imgB * 256 * 784 + h * 28 + w;
        float vals[8];
        float s = 0.f, sq = 0.f;
#pragma unroll
        for (int j = 0; j < 8; ++j) {
            int c = lane + j * 32;
            float v = xb[(size_t)c * 784];
            vals[j] = v; s += v; sq += v * v;
        }
#pragma unroll
        for (int off = 16; off >= 1; off >>= 1) {
            s  += __shfl_xor(s,  off, 32);
            sq += __shfl_xor(sq, off, 32);
        }
        float mu   = s * (1.f / 256.f);
        float var  = sq * (1.f / 256.f) - mu * mu;
        float rstd = rsqrtf(var + 1e-5f);
        ushort_t* orow = xln + ((size_t)win * MPAD + p) * 256;
#pragma unroll
        for (int j = 0; j < 8; ++j) {
            int c = lane + j * 32;
            orow[c] = bfbits((vals[j] - mu) * rstd * pre_w[c] + pre_b[c]);
        }
    }
    for (int r = NPOS; r < MPAD; ++r)
        xln[((size_t)win * MPAD + r) * 256 + threadIdx.x] = 0;
}

// ---- Kernel 2: fused QK(512)+V(1024) GEMM, M=64/window, N-tile=128, K=256
__global__ __launch_bounds__(256) void k_qkv(const ushort_t* __restrict__ xln,
                                             const ushort_t* __restrict__ wqkv,
                                             const float* __restrict__ qk_b,
                                             const float* __restrict__ v_b,
                                             ushort_t* __restrict__ qbuf,
                                             ushort_t* __restrict__ kbuf,
                                             ushort_t* __restrict__ vabuf,
                                             ushort_t* __restrict__ x2) {
    __shared__ __attribute__((aligned(16))) ushort_t Alds[64 * 128];
    __shared__ __attribute__((aligned(16))) ushort_t Wlds[128 * 128];
    int win   = blockIdx.x / 12;
    int nbase = (blockIdx.x % 12) * 128;
    int wave = threadIdx.x >> 5, lane = threadIdx.x & 31;
    int mt = wave & 3, nh = wave >> 2;
    unsigned aLds = (unsigned)(size_t)(void*)Alds;
    unsigned wLds = (unsigned)(size_t)(void*)Wlds;
    v8f acc[4] = {};
    for (int kc = 0; kc < 2; ++kc) {
        int kb0 = kc * 128;
        for (int i = threadIdx.x; i < 1024; i += 256) {          // A: 64x128
            int r = i >> 4, c = (i & 15) << 3;
            async_g2l_b128(aLds + (unsigned)(((r << 7) + c) * 2),
                           xln + ((size_t)win * MPAD + r) * 256 + kb0 + c);
        }
        for (int i = threadIdx.x; i < 2048; i += 256) {          // W: 128x128
            int r = i >> 4, c = (i & 15) << 3;
            async_g2l_b128(wLds + (unsigned)(((r << 7) + c) * 2),
                           wqkv + (size_t)(nbase + r) * 256 + kb0 + c);
        }
        async_wait0();
        __syncthreads();
#pragma unroll
        for (int ks = 0; ks < 4; ++ks) {
            v16bf a = load_a_frag(Alds, mt * 16, ks * 32, 128);
#pragma unroll
            for (int t = 0; t < 4; ++t) {
                v16bf b = load_b_frag_t(Wlds, nh * 64 + t * 16, ks * 32, 128);
                acc[t] = wmma_bf16(a, b, acc[t]);
            }
        }
        __syncthreads();
    }
    int grp = lane >> 4, nl = lane & 15;
#pragma unroll
    for (int t = 0; t < 4; ++t) {
#pragma unroll
        for (int r = 0; r < 8; ++r) {
            int m  = mt * 16 + grp * 8 + r;
            int ng = nbase + nh * 64 + t * 16 + nl;
            float val = acc[t][r];
            if (ng < 512) {                       // q / k
                val += qk_b[ng];
                int s = ng >> 8, head = (ng & 255) >> 5, d = ng & 31;
                ushort_t* dst = (s == 0) ? qbuf : kbuf;
                dst[(((size_t)win * 8 + head) * MPAD + m) * 32 + d] = bfbits(val);
            } else {
                int o = ng - 512;                 // channel within DIM_EXP
                val += v_b[o];
                if (o < 512) {                    // v_attn
                    int head = o >> 6, d = o & 63;
                    vabuf[(((size_t)win * 8 + head) * MPAD + m) * 64 + d] = bfbits(val);
                } else if (m < NPOS) {            // v_idle -> GELU -> proj input
                    x2[((size_t)win * MPAD + m) * 1024 + o] = bfbits(gelu_exact(val));
                }
            }
        }
    }
}

// ---- Kernel 3: attention per (window, head)
__global__ __launch_bounds__(256) void k_attn(const ushort_t* __restrict__ qbuf,
                                              const ushort_t* __restrict__ kbuf,
                                              const ushort_t* __restrict__ vabuf,
                                              const float* __restrict__ rpb,
                                              ushort_t* __restrict__ attn_out) {
    __shared__ __attribute__((aligned(16))) ushort_t qs[64 * 32];
    __shared__ __attribute__((aligned(16))) ushort_t ksm[64 * 32];
    __shared__ __attribute__((aligned(16))) ushort_t vas[64 * 64];
    __shared__ float    Sm[64 * 64];
    __shared__ ushort_t Pm[64 * 64];
    int win = blockIdx.x >> 3, head = blockIdx.x & 7;
    size_t base32 = ((size_t)win * 8 + head) * MPAD * 32;
    size_t base64 = ((size_t)win * 8 + head) * MPAD * 64;
    unsigned qoff = (unsigned)(size_t)(void*)qs;
    unsigned koff = (unsigned)(size_t)(void*)ksm;
    unsigned voff = (unsigned)(size_t)(void*)vas;
    {
        int i = threadIdx.x;                 // 256 chunks of 16B each for q,k
        async_g2l_b128(qoff + (unsigned)(i * 16), qbuf + base32 + (size_t)i * 8);
        async_g2l_b128(koff + (unsigned)(i * 16), kbuf + base32 + (size_t)i * 8);
        async_g2l_b128(voff + (unsigned)(i * 16), vabuf + base64 + (size_t)i * 8);
        async_g2l_b128(voff + (unsigned)((i + 256) * 16),
                       vabuf + base64 + (size_t)(i + 256) * 8);
    }
    async_wait0();
    __syncthreads();
    int wave = threadIdx.x >> 5, lane = threadIdx.x & 31;
    int grp = lane >> 4, nl = lane & 15;
    const float scale = 0.17677669529663687f;   // 32^-0.5
    const float* rpbh = rpb + (size_t)(win & 15) * 169 * 8 + head;
    for (int tt = wave; tt < 16; tt += 8) {
        int it = tt >> 2, jt = tt & 3;
        v16bf a = load_a_frag(qs, it * 16, 0, 32);
        v16bf b = load_b_frag_t(ksm, jt * 16, 0, 32);
        v8f c = {};
        c = wmma_bf16(a, b, c);
#pragma unroll
        for (int r = 0; r < 8; ++r) {
            int m = it * 16 + grp * 8 + r;
            int n = jt * 16 + nl;
            float v;
            if (n >= NPOS)      v = -1e30f;
            else if (m >= NPOS) v = 0.f;
            else {
                int dy = (m / 7) - (n / 7) + 6;
                int dx = (m % 7) - (n % 7) + 6;
                v = c[r] * scale + rpbh[(size_t)(dy * 13 + dx) * 8];
            }
            Sm[m * 64 + n] = v;
        }
    }
    __syncthreads();
    if (threadIdx.x < 64) {
        int row = threadIdx.x;
        float mx = -1e30f;
        for (int j = 0; j < 64; ++j) mx = fmaxf(mx, Sm[row * 64 + j]);
        float s = 0.f;
        for (int j = 0; j < 64; ++j) {
            float ev = expf(Sm[row * 64 + j] - mx);
            Sm[row * 64 + j] = ev; s += ev;
        }
        float inv = 1.f / s;
        for (int j = 0; j < 64; ++j) Pm[row * 64 + j] = bfbits(Sm[row * 64 + j] * inv);
    }
    __syncthreads();
    for (int tt = wave; tt < 16; tt += 8) {
        int it = tt >> 2, dt = tt & 3;
        v8f c = {};
#pragma unroll
        for (int ks = 0; ks < 2; ++ks) {
            v16bf a = load_a_frag(Pm, it * 16, ks * 32, 64);
            v16bf b = load_b_frag_n(vas, dt * 16, ks * 32, 64);
            c = wmma_bf16(a, b, c);
        }
#pragma unroll
        for (int r = 0; r < 8; ++r) {
            int m = it * 16 + grp * 8 + r;
            if (m < NPOS) {
                int d = dt * 16 + nl;
                attn_out[((size_t)win * NPOS + m) * 512 + head * 64 + d] = bfbits(c[r]);
            }
        }
    }
}

// ---- Kernel 4: post channel-LN over 512 attn channels + GELU -> x2 cols 0..511
__global__ __launch_bounds__(256) void k_postln(const ushort_t* __restrict__ attn_out,
                                                const float* __restrict__ post_w,
                                                const float* __restrict__ post_b,
                                                ushort_t* __restrict__ x2) {
    int wave = threadIdx.x >> 5, lane = threadIdx.x & 31;
    size_t row = (size_t)blockIdx.x * 8 + wave;
    if (row >= (size_t)MTOT) return;
    const ushort_t* src = attn_out + row * 512;
    float vals[16]; float s = 0.f, sq = 0.f;
#pragma unroll
    for (int j = 0; j < 16; ++j) {
        float v = bf2f(src[lane + j * 32]);
        vals[j] = v; s += v; sq += v * v;
    }
#pragma unroll
    for (int off = 16; off >= 1; off >>= 1) {
        s  += __shfl_xor(s,  off, 32);
        sq += __shfl_xor(sq, off, 32);
    }
    float mu   = s * (1.f / 512.f);
    float var  = sq * (1.f / 512.f) - mu * mu;
    float rstd = rsqrtf(var + 1e-5f);
    int winr = (int)(row / NPOS), p = (int)(row % NPOS);
    ushort_t* dst = x2 + ((size_t)winr * MPAD + p) * 1024;
#pragma unroll
    for (int j = 0; j < 16; ++j) {
        int c = lane + j * 32;
        float v = (vals[j] - mu) * rstd * post_w[c] + post_b[c];
        dst[c] = bfbits(gelu_exact(v));
    }
}

// ---- Kernel 5: proj GEMM (K=1024 -> N=256) + bias + residual + roll(2,2)
__global__ __launch_bounds__(256) void k_proj(const ushort_t* __restrict__ x2,
                                              const ushort_t* __restrict__ wproj,
                                              const float* __restrict__ proj_b,
                                              const float* __restrict__ x,
                                              float* __restrict__ out) {
    __shared__ __attribute__((aligned(16))) ushort_t Alds[64 * 128];
    __shared__ __attribute__((aligned(16))) ushort_t Wlds[128 * 128];
    int win   = blockIdx.x >> 1;
    int nbase = (blockIdx.x & 1) * 128;
    int wave = threadIdx.x >> 5, lane = threadIdx.x & 31;
    int mt = wave & 3, nh = wave >> 2;
    unsigned aLds = (unsigned)(size_t)(void*)Alds;
    unsigned wLds = (unsigned)(size_t)(void*)Wlds;
    v8f acc[4] = {};
    for (int kc = 0; kc < 8; ++kc) {
        int kb0 = kc * 128;
        for (int i = threadIdx.x; i < 1024; i += 256) {          // A: 64x128
            int r = i >> 4, c = (i & 15) << 3;
            async_g2l_b128(aLds + (unsigned)(((r << 7) + c) * 2),
                           x2 + ((size_t)win * MPAD + r) * 1024 + kb0 + c);
        }
        for (int i = threadIdx.x; i < 2048; i += 256) {          // W: 128x128
            int r = i >> 4, c = (i & 15) << 3;
            async_g2l_b128(wLds + (unsigned)(((r << 7) + c) * 2),
                           wproj + (size_t)(nbase + r) * 1024 + kb0 + c);
        }
        async_wait0();
        __syncthreads();
#pragma unroll
        for (int ks = 0; ks < 4; ++ks) {
            v16bf a = load_a_frag(Alds, mt * 16, ks * 32, 128);
#pragma unroll
            for (int t = 0; t < 4; ++t) {
                v16bf b = load_b_frag_t(Wlds, nh * 64 + t * 16, ks * 32, 128);
                acc[t] = wmma_bf16(a, b, acc[t]);
            }
        }
        __syncthreads();
    }
    int grp = lane >> 4, nl = lane & 15;
    int imgB = win >> 4, wIdx = win & 15;
    int wy = wIdx >> 2, wx = wIdx & 3;
#pragma unroll
    for (int t = 0; t < 4; ++t) {
#pragma unroll
        for (int r = 0; r < 8; ++r) {
            int m = mt * 16 + grp * 8 + r;
            if (m >= NPOS) continue;
            int o  = nbase + nh * 64 + t * 16 + nl;
            int py = m / 7, px = m % 7;
            int h = py * 4 + wy, w = px * 4 + wx;
            float val = acc[t][r] + proj_b[o]
                      + x[(((size_t)imgB * 256 + o) * 28 + h) * 28 + w];
            int hh = h + 2; if (hh >= 28) hh -= 28;
            int ww = w + 2; if (ww >= 28) ww -= 28;
            out[(((size_t)imgB * 256 + o) * 28 + hh) * 28 + ww] = val;
        }
    }
}

extern "C" void kernel_launch(void* const* d_in, const int* in_sizes, int n_in,
                              void* d_out, int out_size, void* d_ws, size_t ws_size,
                              hipStream_t stream) {
    (void)in_sizes; (void)n_in; (void)out_size; (void)ws_size;
    const float* x      = (const float*)d_in[0];
    const float* pre_w  = (const float*)d_in[1];
    const float* pre_b  = (const float*)d_in[2];
    const float* qk_w   = (const float*)d_in[3];
    const float* qk_b   = (const float*)d_in[4];
    const float* v_w    = (const float*)d_in[5];
    const float* v_b    = (const float*)d_in[6];
    const float* post_w = (const float*)d_in[7];
    const float* post_b = (const float*)d_in[8];
    const float* proj_w = (const float*)d_in[9];
    const float* proj_b = (const float*)d_in[10];
    const float* rpb    = (const float*)d_in[11];

    char* ws = (char*)d_ws;
    size_t off = 0;
    ushort_t* xln   = (ushort_t*)(ws + off); off += (size_t)NWIN * MPAD * 256 * 2;
    ushort_t* qbuf  = (ushort_t*)(ws + off); off += (size_t)NWIN * 8 * MPAD * 32 * 2;
    ushort_t* kbuf  = (ushort_t*)(ws + off); off += (size_t)NWIN * 8 * MPAD * 32 * 2;
    ushort_t* vabuf = (ushort_t*)(ws + off); off += (size_t)NWIN * 8 * MPAD * 64 * 2;
    ushort_t* aout  = (ushort_t*)(ws + off); off += (size_t)MTOT * 512 * 2;
    ushort_t* x2    = (ushort_t*)(ws + off); off += (size_t)NWIN * MPAD * 1024 * 2;
    ushort_t* wqkv  = (ushort_t*)(ws + off); off += (size_t)1536 * 256 * 2;
    ushort_t* wproj = (ushort_t*)(ws + off); off += (size_t)256 * 1024 * 2;
    float* out = (float*)d_out;

    // one-time (per launch) weight conversion + pad-row zeroing
    k_cvt<<<(512 * 256 + 255) / 256, 256, 0, stream>>>(qk_w, wqkv, 512 * 256);
    k_cvt<<<(1024 * 256 + 255) / 256, 256, 0, stream>>>(v_w, wqkv + (size_t)512 * 256, 1024 * 256);
    k_cvt<<<(256 * 1024 + 255) / 256, 256, 0, stream>>>(proj_w, wproj, 256 * 1024);
    k_zero_pad<<<(NWIN * 15 * 128) / 256, 256, 0, stream>>>(x2);

    k_ln    <<<NWIN,      256, 0, stream>>>(x, pre_w, pre_b, xln);
    k_qkv   <<<NWIN * 12, 256, 0, stream>>>(xln, wqkv, qk_b, v_b,
                                            qbuf, kbuf, vabuf, x2);
    k_attn  <<<NWIN * 8,  256, 0, stream>>>(qbuf, kbuf, vabuf, rpb, aout);
    k_postln<<<MTOT / 8,  256, 0, stream>>>(aout, post_w, post_b, x2);
    k_proj  <<<NWIN * 2,  256, 0, stream>>>(x2, wproj, proj_b, x, out);
}
